// EGNN_Sparse_Network_8546984919108
// MI455X (gfx1250) — compile-verified, matrix operand order checked by reading, not run on hardware
//
#include <hip/hip_runtime.h>
#include <hip/hip_bf16.h>

// ---------------------------------------------------------------------------
// EGNN sparse network, 2 layers, for gfx1250 (MI455X).
// Edge MLP + coord gate via v_wmma_f32_16x16x32_f16 (f16 in, f32 accum).
// Weights pre-packed into WMMA fragment-major layout (zero-padded K/N) ->
// B-fragments are clause-batched global_load_b128 pairs. Biases zero-padded
// so C-stores are branchless; SiLU uses hardware v_exp/v_rcp.
// ---------------------------------------------------------------------------

typedef __attribute__((ext_vector_type(16))) _Float16 v16h;
typedef __attribute__((ext_vector_type(8)))  _Float16 v8h;
typedef __attribute__((ext_vector_type(8)))  float    v8f;

#define Nn   50000
#define Ee   800000
#define Pp   3
#define Ff   64
#define Mm   32
#define Ll   2
#define XROW 67          // P + F
#define EIN  129         // 2F + 1
#define EPSf 1e-8f

// ----------------------------- WMMA helpers --------------------------------
// A: 16xK tile in LDS (row-major f16, leading dim ld, 16B-aligned rows).
// CDNA5 16-bit A layout: lane l -> row (l&15); half h=(l>>4):
// elems 0..7 = K k0+h*8+i, elems 8..15 = K k0+16+h*8+i -> ds_load_b128 x2.
__device__ __forceinline__ v16h frag_A_lds(const _Float16* a, int ld, int k0, int lane) {
    const int r = lane & 15;
    const int h = (lane >> 4) & 1;
    const _Float16* p = a + r * ld + k0 + h * 8;
    v8h lo = *(const v8h*)(p);
    v8h hi = *(const v8h*)(p + 16);
    return __builtin_shufflevector(lo, hi, 0, 1, 2, 3, 4, 5, 6, 7,
                                   8, 9, 10, 11, 12, 13, 14, 15);
}

// B: packed fragment-major weights; per-lane chunk is 32B -> global_load_b128 x2.
__device__ __forceinline__ v16h frag_B_pk(const _Float16* __restrict__ Wp,
                                          int kt, int nt, int NT, int lane) {
    const _Float16* p = Wp + ((((size_t)kt * NT + nt) * 32 + lane) << 4);
    v8h lo = *(const v8h*)(p);
    v8h hi = *(const v8h*)(p + 8);
    return __builtin_shufflevector(lo, hi, 0, 1, 2, 3, 4, 5, 6, 7,
                                   8, 9, 10, 11, 12, 13, 14, 15);
}

__device__ __forceinline__ v8f wmma_f16(v16h a, v16h b, v8f c) {
    return __builtin_amdgcn_wmma_f32_16x16x32_f16(false, a, false, b, (short)0, c,
                                                  false, false);
}

// Fast SiLU: x * rcp(1+exp(-x)) -> v_exp_f32 + v_rcp_f32 + v_mul (no IEEE div).
__device__ __forceinline__ float silu(float x) {
    return x * __builtin_amdgcn_rcpf(1.f + __expf(-x));
}

// C/D layout: lane l holds col n0+(l&15); VGPR r holds row r + 8*(l>>4).
// Branchless: bias is zero-padded; packed-weight pad cols give silu(0)=0.
__device__ __forceinline__ void store_C_silu_lds(v8f c, const float* __restrict__ bias,
                                                 _Float16* dst, int ld,
                                                 int n0, int lane) {
    const int n = n0 + (lane & 15);
    const int rbase = ((lane >> 4) & 1) * 8;
    const float b = bias[n];
#pragma unroll
    for (int r = 0; r < 8; ++r) {
        dst[(rbase + r) * ld + n] = (_Float16)silu(c[r] + b);
    }
}

// ----------------------------- utility kernels -----------------------------
// Pack a (kvalid x nvalid) row-major f32 weight matrix into fragment-major f16:
// tiles (kt,nt), 32 lanes x 16 halves, padding zero-filled.
__global__ void pack_weights(const float* __restrict__ W, _Float16* __restrict__ Wp,
                             int kvalid, int nvalid, int NT, int total) {
    int idx = blockIdx.x * blockDim.x + threadIdx.x;
    if (idx >= total) return;
    const int i    = idx & 15;
    const int lane = (idx >> 4) & 31;
    const int tile = idx >> 9;
    const int nt   = tile % NT;
    const int kt   = tile / NT;
    const int n    = nt * 16 + (lane & 15);
    const int h    = (lane >> 4) & 1;
    const int k    = kt * 32 + ((i < 8) ? (h * 8 + i) : (16 + h * 8 + (i - 8)));
    float v = 0.f;
    if (k < kvalid && n < nvalid) v = W[(size_t)k * nvalid + n];
    Wp[idx] = (_Float16)v;
}

__global__ void pad_bias(const float* __restrict__ s, float* __restrict__ d,
                         int n, int npad) {
    int i = blockIdx.x * blockDim.x + threadIdx.x;
    if (i < npad) d[i] = (i < n) ? s[i] : 0.f;
}

__global__ void zero_f32(float* __restrict__ p, int n) {
    int i = blockIdx.x * blockDim.x + threadIdx.x;
    if (i < n) p[i] = 0.f;
}

// ------------------------------- edge kernel -------------------------------
// 64 threads = 2 waves; each wave owns one 16-edge tile (E/16 = 50000 tiles).
__global__ __launch_bounds__(64)
void egnn_edge_kernel(const float* __restrict__ x,
                      const int* __restrict__ srcI, const int* __restrict__ dstI,
                      const _Float16* __restrict__ eW1p, const float* __restrict__ eb1p,
                      const _Float16* __restrict__ eW2p, const float* __restrict__ eb2,
                      const _Float16* __restrict__ cW1p, const float* __restrict__ cb1,
                      const float* __restrict__ cW2,    const float* __restrict__ cb2,
                      const float* __restrict__ cscale,
                      float* __restrict__ m_i, float* __restrict__ mhat) {
    __shared__ alignas(16) _Float16 s_ein[2][16][160];  // e_in padded K: 129 -> 160
    __shared__ alignas(16) _Float16 s_h  [2][16][288];  // hidden padded: 258 -> 288
    __shared__ alignas(16) _Float16 s_m  [2][16][32];   // m_ij
    __shared__ alignas(16) _Float16 s_c1 [2][16][128];  // coord-MLP hidden
    __shared__ int s_dst[2][16];
    __shared__ int s_src[2][16];

    const int lane = threadIdx.x & 31;
    const int w    = threadIdx.x >> 5;
    const int tile = blockIdx.x * 2 + w;    // always < 50000

    // ---- stage 1: indices, coords, rel, dist (lanes 0..15, kept in regs) --
    float relx = 0.f, rely = 0.f, relz = 0.f, dist = 0.f;
    int dreg = 0;
    if (lane < 16) {
        const int e = tile * 16 + lane;
        const int s = srcI[e];
        const int d = dstI[e];
        dreg = d;
        s_src[w][lane] = s;
        s_dst[w][lane] = d;
        relx = x[(size_t)s * XROW + 0] - x[(size_t)d * XROW + 0];
        rely = x[(size_t)s * XROW + 1] - x[(size_t)d * XROW + 1];
        relz = x[(size_t)s * XROW + 2] - x[(size_t)d * XROW + 2];
        dist = relx * relx + rely * rely + relz * relz;
        s_ein[w][lane][128] = (_Float16)dist;
    }
    __syncthreads();

    // ---- stage 2: gather features, zero pads ------------------------------
#pragma unroll 1
    for (int r = 0; r < 16; ++r) {
        const int d = s_dst[w][r];
        const int s = s_src[w][r];
#pragma unroll
        for (int it = 0; it < 2; ++it) {
            const int c = lane + it * 32;                       // 0..63
            s_ein[w][r][c]      = (_Float16)x[(size_t)d * XROW + Pp + c];
            s_ein[w][r][64 + c] = (_Float16)x[(size_t)s * XROW + Pp + c];
        }
        if (lane >= 1) s_ein[w][r][128 + lane] = (_Float16)0.f; // cols 129..159
        if (lane < 16) s_h[w][r][272 + lane]   = (_Float16)0.f; // cols 272..287
    }
    __syncthreads();

    // ---- GEMM1: h = silu(e_in @ eW1 + eb1); 16x258, K=129 (5 kt, 17 nt) ---
#pragma unroll 1
    for (int nt = 0; nt < 17; ++nt) {
        v8f c = {};
#pragma unroll
        for (int kt = 0; kt < 5; ++kt) {
            v16h a = frag_A_lds(&s_ein[w][0][0], 160, kt * 32, lane);
            v16h b = frag_B_pk(eW1p, kt, nt, 17, lane);
            c = wmma_f16(a, b, c);
        }
        store_C_silu_lds(c, eb1p, &s_h[w][0][0], 288, nt * 16, lane);
    }
    __syncthreads();

    // ---- GEMM2: m = silu(h @ eW2 + eb2); 16x32, K=258 (9 kt, 2 nt) --------
#pragma unroll 1
    for (int nt = 0; nt < 2; ++nt) {
        v8f c = {};
#pragma unroll
        for (int kt = 0; kt < 9; ++kt) {
            v16h a = frag_A_lds(&s_h[w][0][0], 288, kt * 32, lane);
            v16h b = frag_B_pk(eW2p, kt, nt, 2, lane);
            c = wmma_f16(a, b, c);
        }
        store_C_silu_lds(c, eb2, &s_m[w][0][0], 32, nt * 16, lane);
    }
    __syncthreads();

    // ---- GEMM3: c1 = silu(m @ cW1 + cb1); 16x128, K=32 (1 kt, 8 nt) -------
#pragma unroll 1
    for (int nt = 0; nt < 8; ++nt) {
        v16h a = frag_A_lds(&s_m[w][0][0], 32, 0, lane);
        v16h b = frag_B_pk(cW1p, 0, nt, 8, lane);
        v8f c = {};
        c = wmma_f16(a, b, c);
        store_C_silu_lds(c, cb1, &s_c1[w][0][0], 128, nt * 16, lane);
    }

    // ---- scatter m_ij into m_i (lane = channel, loop rows) ----------------
#pragma unroll 1
    for (int r = 0; r < 16; ++r) {
        atomicAdd(&m_i[(size_t)s_dst[w][r] * Mm + lane], (float)s_m[w][r][lane]);
    }
    __syncthreads();

    // ---- GEMM4 (matvec) + tanh gate + coordinate scatter ------------------
    if (lane < 16) {
        float acc = cb2[0];
#pragma unroll
        for (int k = 0; k < 128; ++k) acc += (float)s_c1[w][lane][k] * cW2[k];
        const float cw  = tanhf(acc);
        const float nrm = sqrtf(dist);
        const float g   = cw * cscale[0] / fmaxf(nrm, EPSf);
        atomicAdd(&mhat[(size_t)dreg * 3 + 0], g * relx);
        atomicAdd(&mhat[(size_t)dreg * 3 + 1], g * rely);
        atomicAdd(&mhat[(size_t)dreg * 3 + 2], g * relz);
    }
}

// ------------------------------- node kernel -------------------------------
// 64 threads = 2 waves; each wave owns one 16-node tile (N/16 = 3125 tiles).
__global__ __launch_bounds__(64)
void egnn_node_kernel(const float* __restrict__ x,
                      const float* __restrict__ m_i, const float* __restrict__ mhat,
                      const _Float16* __restrict__ nW1p, const float* __restrict__ nb1,
                      const _Float16* __restrict__ nW2p, const float* __restrict__ nb2,
                      float* __restrict__ xout) {
    __shared__ alignas(16) _Float16 s_nin[2][16][96];   // concat(feats, m_i)
    __shared__ alignas(16) _Float16 s_t  [2][16][128];  // hidden

    const int lane  = threadIdx.x & 31;
    const int w     = threadIdx.x >> 5;
    int tile        = blockIdx.x * 2 + w;
    const bool valid = (tile < Nn / 16);
    if (!valid) tile = Nn / 16 - 1;         // clamp; no early return (barriers)

    // ---- gather n_in ------------------------------------------------------
#pragma unroll 1
    for (int r = 0; r < 16; ++r) {
        const size_t node = (size_t)tile * 16 + r;
#pragma unroll
        for (int it = 0; it < 3; ++it) {
            const int c = lane + it * 32;   // 0..95
            const float v = (c < Ff) ? x[node * XROW + Pp + c]
                                     : m_i[node * Mm + (c - Ff)];
            s_nin[w][r][c] = (_Float16)v;
        }
    }
    __syncthreads();

    // ---- GEMM1: t = silu(n_in @ nW1 + nb1); 16x128, K=96 (3 kt, 8 nt) -----
#pragma unroll 1
    for (int nt = 0; nt < 8; ++nt) {
        v8f c = {};
#pragma unroll
        for (int kt = 0; kt < 3; ++kt) {
            v16h a = frag_A_lds(&s_nin[w][0][0], 96, kt * 32, lane);
            v16h b = frag_B_pk(nW1p, kt, nt, 8, lane);
            c = wmma_f16(a, b, c);
        }
        store_C_silu_lds(c, nb1, &s_t[w][0][0], 128, nt * 16, lane);
    }
    __syncthreads();

    // ---- GEMM2: upd = t @ nW2 + nb2; residual feature write (4 kt, 4 nt) --
#pragma unroll 1
    for (int nt = 0; nt < 4; ++nt) {
        v8f c = {};
#pragma unroll
        for (int kt = 0; kt < 4; ++kt) {
            v16h a = frag_A_lds(&s_t[w][0][0], 128, kt * 32, lane);
            v16h b = frag_B_pk(nW2p, kt, nt, 4, lane);
            c = wmma_f16(a, b, c);
        }
        const int n = nt * 16 + (lane & 15);
        const int rbase = ((lane >> 4) & 1) * 8;
        const float b = nb2[n];
        if (valid) {
#pragma unroll
            for (int r = 0; r < 8; ++r) {
                const size_t node = (size_t)tile * 16 + rbase + r;
                xout[node * XROW + Pp + n] = x[node * XROW + Pp + n] + c[r] + b;
            }
        }
    }

    // ---- coordinate residual ----------------------------------------------
    if (valid && lane < 16) {
        const size_t node = (size_t)tile * 16 + lane;
#pragma unroll
        for (int p = 0; p < Pp; ++p)
            xout[node * XROW + p] = x[node * XROW + p] + mhat[node * 3 + p];
    }
}

// ------------------------------- host launch -------------------------------
static inline size_t ktiles(size_t k) { return (k + 31) / 32; }
static inline size_t ntiles(size_t n) { return (n + 15) / 16; }
static inline size_t packed_sz(size_t k, size_t n) { return ktiles(k) * ntiles(n) * 512; }

extern "C" void kernel_launch(void* const* d_in, const int* in_sizes, int n_in,
                              void* d_out, int out_size, void* d_ws, size_t ws_size,
                              hipStream_t stream) {
    const float* x_in  = (const float*)d_in[0];
    const int*   ei    = (const int*)d_in[1];
    const float* eW1   = (const float*)d_in[2];
    const float* eb1   = (const float*)d_in[3];
    const float* eW2   = (const float*)d_in[4];
    const float* eb2   = (const float*)d_in[5];
    const float* cW1   = (const float*)d_in[6];
    const float* cb1   = (const float*)d_in[7];
    const float* cW2   = (const float*)d_in[8];
    const float* cb2   = (const float*)d_in[9];
    const float* nW1   = (const float*)d_in[10];
    const float* nb1   = (const float*)d_in[11];
    const float* nW2   = (const float*)d_in[12];
    const float* nb2   = (const float*)d_in[13];
    const float* cscal = (const float*)d_in[14];

    // workspace layout (f32 region, packed f16 weights, padded eb1)
    float* m_i  = (float*)d_ws;                          // N*32 f32
    float* mhat = m_i + (size_t)Nn * Mm;                 // N*3  f32
    float* xbuf = mhat + (size_t)Nn * 3;                 // N*67 f32
    _Float16* wp = (_Float16*)(xbuf + (size_t)Nn * XROW);

    const size_t peW1 = packed_sz(EIN, 258);   // 5*17*512
    const size_t peW2 = packed_sz(258, 32);    // 9*2*512
    const size_t pcW1 = packed_sz(32, 128);    // 1*8*512
    const size_t pnW1 = packed_sz(96, 128);    // 3*8*512
    const size_t pnW2 = packed_sz(128, 64);    // 4*4*512
    const size_t WL   = peW1 + peW2 + pcW1 + pnW1 + pnW2;

    float* eb1pad = (float*)(wp + (size_t)Ll * WL);      // L x 272 f32 (4B aligned)

    const size_t szeW1 = (size_t)EIN * 258, szeW2 = 258 * 32, szcW1 = 32 * 128,
                 sznW1 = 96 * 128, sznW2 = 128 * 64;

    // pack all weights + pad eb1 (both layers)
    for (int l = 0; l < Ll; ++l) {
        _Float16* base = wp + (size_t)l * WL;
        struct { const float* s; _Float16* d; int k, n; } jobs[5] = {
            { eW1 + l * szeW1, base,                        EIN, 258 },
            { eW2 + l * szeW2, base + peW1,                 258, 32 },
            { cW1 + l * szcW1, base + peW1 + peW2,          32, 128 },
            { nW1 + l * sznW1, base + peW1 + peW2 + pcW1,   96, 128 },
            { nW2 + l * sznW2, base + peW1 + peW2 + pcW1 + pnW1, 128, 64 },
        };
        for (int j = 0; j < 5; ++j) {
            int NT = (int)ntiles(jobs[j].n);
            int total = (int)packed_sz(jobs[j].k, jobs[j].n);
            pack_weights<<<(total + 255) / 256, 256, 0, stream>>>(
                jobs[j].s, jobs[j].d, jobs[j].k, jobs[j].n, NT, total);
        }
        pad_bias<<<(272 + 255) / 256, 256, 0, stream>>>(eb1 + (size_t)l * 258,
                                                        eb1pad + (size_t)l * 272,
                                                        258, 272);
    }

    for (int l = 0; l < Ll; ++l) {
        const float* x_cur = (l == 0) ? x_in : xbuf;
        float*       x_out = (l == Ll - 1) ? (float*)d_out : xbuf;
        _Float16* base = wp + (size_t)l * WL;
        const _Float16* eW1p = base;
        const _Float16* eW2p = base + peW1;
        const _Float16* cW1p = base + peW1 + peW2;
        const _Float16* nW1p = base + peW1 + peW2 + pcW1;
        const _Float16* nW2p = base + peW1 + peW2 + pcW1 + pnW1;

        // zero segment-sum accumulators (m_i and mhat are contiguous)
        {
            int n = Nn * (Mm + 3);
            zero_f32<<<(n + 255) / 256, 256, 0, stream>>>(m_i, n);
        }

        egnn_edge_kernel<<<Ee / 16 / 2, 64, 0, stream>>>(
            x_cur, ei, ei + Ee,
            eW1p, eb1pad + (size_t)l * 272,
            eW2p, eb2 + (size_t)l * 32,
            cW1p, cb1 + (size_t)l * 128,
            cW2 + (size_t)l * 128, cb2 + l,
            cscal + l, m_i, mhat);

        egnn_node_kernel<<<(Nn / 16 + 1) / 2, 64, 0, stream>>>(
            x_cur, m_i, mhat,
            nW1p, nb1 + (size_t)l * 128,
            nW2p, nb2 + (size_t)l * 64,
            x_out);
    }
}
